// GNNModel_4217657884943
// MI455X (gfx1250) — matrix operand order repaired and checked by reading, not compiled, not run on hardware
//
#include <hip/hip_runtime.h>
#include <math.h>

typedef __attribute__((ext_vector_type(16))) __bf16 v16bf;
typedef __attribute__((ext_vector_type(8)))  float  v8f;

// ---------------------------------------------------------------------------
// Degree / normalization
// ---------------------------------------------------------------------------
__global__ void k_fill1(float* __restrict__ d, int n) {
    int i = blockIdx.x * blockDim.x + threadIdx.x;
    if (i < n) d[i] = 1.0f;                       // self-loop contributes 1
}

__global__ void k_deg(const long long* __restrict__ col, float* __restrict__ deg, int E) {
    int e = blockIdx.x * blockDim.x + threadIdx.x;
    if (e < E) atomicAdd(&deg[(long)col[e]], 1.0f);
}

__global__ void k_rsqrt(float* __restrict__ d, int n) {
    int i = blockIdx.x * blockDim.x + threadIdx.x;
    if (i < n) d[i] = rsqrtf(d[i]);
}

// ---------------------------------------------------------------------------
// Pack 128x128 f32 weight (row-major, K x N) into per-lane WMMA B layout:
// pW[((t*4+s)*32 + lane)*16 + i] = bf16( W[(s*32 + (i&8)*2 + (lane>>4)*8 + (i&7))*128 + t*16 + (lane&15)] )
// so the GEMM does one contiguous v16bf (32B) load per lane per K-step.
// ---------------------------------------------------------------------------
__global__ void k_pack_w128(const float* __restrict__ W, __bf16* __restrict__ pW) {
    int idx = blockIdx.x * blockDim.x + threadIdx.x;   // 0..16383
    if (idx >= 128 * 128) return;
    int i = idx & 15;
    int l = (idx >> 4) & 31;
    int s = (idx >> 9) & 3;
    int t = idx >> 11;
    int k = s * 32 + (i & 8) * 2 + (l >> 4) * 8 + (i & 7);
    int n = t * 16 + (l & 15);
    pW[idx] = (__bf16)W[k * 128 + n];
}

// ---------------------------------------------------------------------------
// C[N x 128] = A[N x 128] (f32, converted to bf16 in-register) @ W (packed bf16)
// One block = 256 threads = 8 waves; wave w owns output-feature tile w (16 cols)
// of the same 16 rows. K = 128 handled as 4x v_wmma_f32_16x16x32_bf16.
// ---------------------------------------------------------------------------
__global__ __launch_bounds__(256) void k_gemm128(const float* __restrict__ A,
                                                 const __bf16* __restrict__ pW,
                                                 float* __restrict__ C) {
    const int wave = threadIdx.x >> 5;          // 0..7 : output feature tile
    const int lane = threadIdx.x & 31;
    const int m    = lane & 15;                 // A row within tile
    const int kh   = lane >> 4;                 // K-half select
    const long node0 = (long)blockIdx.x * 16;
    const float* __restrict__ arow = A + (node0 + m) * 128;

    v8f c = {};
#pragma unroll
    for (int s = 0; s < 4; ++s) {
        // A operand, ISA 16-bit 16x32 layout: a[0..7] = K s*32+kh*8.., a[8..15] = K s*32+16+kh*8..
        const float4 a0 = *(const float4*)(arow + s * 32 + kh * 8);
        const float4 a1 = *(const float4*)(arow + s * 32 + kh * 8 + 4);
        const float4 a2 = *(const float4*)(arow + s * 32 + 16 + kh * 8);
        const float4 a3 = *(const float4*)(arow + s * 32 + 16 + kh * 8 + 4);
        v16bf a;
        a[0]  = (__bf16)a0.x; a[1]  = (__bf16)a0.y; a[2]  = (__bf16)a0.z; a[3]  = (__bf16)a0.w;
        a[4]  = (__bf16)a1.x; a[5]  = (__bf16)a1.y; a[6]  = (__bf16)a1.z; a[7]  = (__bf16)a1.w;
        a[8]  = (__bf16)a2.x; a[9]  = (__bf16)a2.y; a[10] = (__bf16)a2.z; a[11] = (__bf16)a2.w;
        a[12] = (__bf16)a3.x; a[13] = (__bf16)a3.y; a[14] = (__bf16)a3.z; a[15] = (__bf16)a3.w;

        const v16bf b = *(const v16bf*)(pW + ((size_t)(wave * 4 + s) * 32 + lane) * 16);

        c = __builtin_amdgcn_wmma_f32_16x16x32_bf16(false, a, false, b,
                                                    (short)0, c, false, false);
    }

    // C/D layout: VGPR j -> row = kh*8 + j, col = lane&15 (within tiles)
    float* __restrict__ crow = C + node0 * 128 + wave * 16 + m;
#pragma unroll
    for (int j = 0; j < 8; ++j)
        crow[(long)(kh * 8 + j) * 128] = c[j];
}

// ---------------------------------------------------------------------------
// agg[i][f] = dinv[i]^2 * xw[i][f] + bias[f]   (full overwrite -> no memset)
// ---------------------------------------------------------------------------
__global__ void k_selfinit(const float* __restrict__ dinv, const float* __restrict__ xw,
                           const float* __restrict__ bias, float* __restrict__ agg, int n) {
    long idx = (long)blockIdx.x * blockDim.x + threadIdx.x;
    if (idx >= (long)n * 128) return;
    long i = idx >> 7;
    int  f = idx & 127;
    float di = dinv[i];
    agg[idx] = di * di * xw[idx] + bias[f];
}

// ---------------------------------------------------------------------------
// Edge scatter: wave per edge, lane handles 4 features (float4 gather + 4 atomics)
// ---------------------------------------------------------------------------
__global__ __launch_bounds__(256) void k_edge(const long long* __restrict__ row,
                                              const long long* __restrict__ col,
                                              const float* __restrict__ dinv,
                                              const float* __restrict__ xw,
                                              float* __restrict__ agg, int E) {
    int e    = (int)(((long)blockIdx.x * blockDim.x + threadIdx.x) >> 5);
    int lane = threadIdx.x & 31;
    if (e >= E) return;
    long r = (long)row[e];
    long c = (long)col[e];
    float w = dinv[r] * dinv[c];
    float4 v = *(const float4*)(xw + r * 128 + lane * 4);
    float* dst = agg + c * 128 + lane * 4;
    atomicAdd(dst + 0, w * v.x);
    atomicAdd(dst + 1, w * v.y);
    atomicAdd(dst + 2, w * v.z);
    atomicAdd(dst + 3, w * v.w);
}

// ---------------------------------------------------------------------------
// ReLU (+ optional LayerNorm over 128 feats), in place. Wave per node (wave32).
// ---------------------------------------------------------------------------
__global__ __launch_bounds__(256) void k_post(float* __restrict__ h,
                                              const float* __restrict__ g,
                                              const float* __restrict__ b,
                                              int doLN, int n) {
    int node = (int)(((long)blockIdx.x * blockDim.x + threadIdx.x) >> 5);
    int lane = threadIdx.x & 31;
    if (node >= n) return;
    float4 v = *(float4*)(h + (long)node * 128 + lane * 4);
    v.x = fmaxf(v.x, 0.f); v.y = fmaxf(v.y, 0.f);
    v.z = fmaxf(v.z, 0.f); v.w = fmaxf(v.w, 0.f);
    if (doLN) {
        float s = v.x + v.y + v.z + v.w;
#pragma unroll
        for (int off = 16; off; off >>= 1) s += __shfl_xor(s, off, 32);
        float mu = s * (1.0f / 128.0f);
        float dx = v.x - mu, dy = v.y - mu, dz = v.z - mu, dw = v.w - mu;
        float q = dx * dx + dy * dy + dz * dz + dw * dw;
#pragma unroll
        for (int off = 16; off; off >>= 1) q += __shfl_xor(q, off, 32);
        float inv = rsqrtf(q * (1.0f / 128.0f) + 1e-5f);
        int f = lane * 4;
        v.x = dx * inv * g[f + 0] + b[f + 0];
        v.y = dy * inv * g[f + 1] + b[f + 1];
        v.z = dz * inv * g[f + 2] + b[f + 2];
        v.w = dw * inv * g[f + 3] + b[f + 3];
    }
    *(float4*)(h + (long)node * 128 + lane * 4) = v;
}

// ---------------------------------------------------------------------------
// MLP head + softmax: one 64-thread block per node.
// hidden = relu(h @ Wp1 + bp1) [64]; logits = hidden @ Wp2 + bp2 [10]; softmax.
// ---------------------------------------------------------------------------
__global__ __launch_bounds__(64) void k_mlp(const float* __restrict__ h,
                                            const float* __restrict__ Wp1,
                                            const float* __restrict__ bp1,
                                            const float* __restrict__ Wp2,
                                            const float* __restrict__ bp2,
                                            float* __restrict__ out, int n) {
    __shared__ float sh[128];
    __shared__ float hid[64];
    __shared__ float logits[10];
    long node = blockIdx.x;
    if (node >= n) return;
    int t = threadIdx.x;
    sh[t]      = h[node * 128 + t];
    sh[t + 64] = h[node * 128 + 64 + t];
    __syncthreads();
    float acc = bp1[t];
#pragma unroll 8
    for (int k = 0; k < 128; ++k) acc = fmaf(sh[k], Wp1[k * 64 + t], acc);
    hid[t] = fmaxf(acc, 0.f);
    __syncthreads();
    if (t < 10) {
        float a = bp2[t];
#pragma unroll 8
        for (int k = 0; k < 64; ++k) a = fmaf(hid[k], Wp2[k * 10 + t], a);
        logits[t] = a;
    }
    __syncthreads();
    if (t == 0) {
        float mx = logits[0];
#pragma unroll
        for (int i = 1; i < 10; ++i) mx = fmaxf(mx, logits[i]);
        float ex[10], ssum = 0.f;
#pragma unroll
        for (int i = 0; i < 10; ++i) { ex[i] = __expf(logits[i] - mx); ssum += ex[i]; }
        float inv = 1.0f / ssum;
#pragma unroll
        for (int i = 0; i < 10; ++i) out[node * 10 + i] = ex[i] * inv;
    }
}

// ---------------------------------------------------------------------------
// Host orchestration
// ---------------------------------------------------------------------------
static inline size_t align256(size_t x) { return (x + 255) & ~(size_t)255; }

extern "C" void kernel_launch(void* const* d_in, const int* in_sizes, int n_in,
                              void* d_out, int out_size, void* d_ws, size_t ws_size,
                              hipStream_t stream) {
    const float*     x    = (const float*)d_in[0];
    const long long* eidx = (const long long*)d_in[1];   // int64 (2, E)
    const float* W0 = (const float*)d_in[2];  const float* b0 = (const float*)d_in[3];
    const float* W1 = (const float*)d_in[4];  const float* b1 = (const float*)d_in[5];
    const float* W2 = (const float*)d_in[6];  const float* b2 = (const float*)d_in[7];
    const float* ln0g = (const float*)d_in[8];  const float* ln0b = (const float*)d_in[9];
    const float* ln1g = (const float*)d_in[10]; const float* ln1b = (const float*)d_in[11];
    const float* Wp1 = (const float*)d_in[12]; const float* bp1 = (const float*)d_in[13];
    const float* Wp2 = (const float*)d_in[14]; const float* bp2 = (const float*)d_in[15];
    float* out = (float*)d_out;

    const int N = in_sizes[0] / 128;
    const int E = in_sizes[1] / 2;
    const long long* row = eidx;
    const long long* col = eidx + E;

    // workspace carve-up
    char* ws = (char*)d_ws;
    size_t off = 0;
    float* dinv = (float*)(ws + off); off = align256(off + (size_t)N * 4);
    float* xw   = (float*)(ws + off); off = align256(off + (size_t)N * 128 * 4);
    float* agg  = (float*)(ws + off); off = align256(off + (size_t)N * 128 * 4);
    __bf16* pW0 = (__bf16*)(ws + off); off = align256(off + 128 * 128 * 2);
    __bf16* pW1 = (__bf16*)(ws + off); off = align256(off + 128 * 128 * 2);
    __bf16* pW2 = (__bf16*)(ws + off); off = align256(off + 128 * 128 * 2);
    (void)ws_size; (void)n_in; (void)out_size;

    const int T = 256;
    dim3 blkN((N + T - 1) / T), blkE((E + T - 1) / T);
    dim3 blkNF(((long)N * 128 + T - 1) / T);
    dim3 blkEdge(((long)E * 32 + T - 1) / T);   // wave per edge
    dim3 blkPost(((long)N * 32 + T - 1) / T);   // wave per node
    dim3 blkGemm(N / 16);                       // 100000/16 = 6250

    // degrees -> dinv
    k_fill1<<<blkN, T, 0, stream>>>(dinv, N);
    k_deg<<<blkE, T, 0, stream>>>(col, dinv, E);
    k_rsqrt<<<blkN, T, 0, stream>>>(dinv, N);

    // pack GCN weights into WMMA B layout (bf16)
    k_pack_w128<<<64, 256, 0, stream>>>(W0, pW0);
    k_pack_w128<<<64, 256, 0, stream>>>(W1, pW1);
    k_pack_w128<<<64, 256, 0, stream>>>(W2, pW2);

    // Layer 0
    k_gemm128<<<blkGemm, T, 0, stream>>>(x, pW0, xw);
    k_selfinit<<<blkNF, T, 0, stream>>>(dinv, xw, b0, agg, N);
    k_edge<<<blkEdge, T, 0, stream>>>(row, col, dinv, xw, agg, E);
    k_post<<<blkPost, T, 0, stream>>>(agg, ln0g, ln0b, 1, N);

    // Layer 1
    k_gemm128<<<blkGemm, T, 0, stream>>>(agg, pW1, xw);
    k_selfinit<<<blkNF, T, 0, stream>>>(dinv, xw, b1, agg, N);
    k_edge<<<blkEdge, T, 0, stream>>>(row, col, dinv, xw, agg, E);
    k_post<<<blkPost, T, 0, stream>>>(agg, ln1g, ln1b, 1, N);

    // Layer 2 (no LayerNorm)
    k_gemm128<<<blkGemm, T, 0, stream>>>(agg, pW2, xw);
    k_selfinit<<<blkNF, T, 0, stream>>>(dinv, xw, b2, agg, N);
    k_edge<<<blkEdge, T, 0, stream>>>(row, col, dinv, xw, agg, E);
    k_post<<<blkPost, T, 0, stream>>>(agg, ln0g, ln0b, 0, N);

    // MLP head + softmax
    k_mlp<<<dim3(N), 64, 0, stream>>>(agg, Wp1, bp1, Wp2, bp2, out, N);
}
